// Qwen3_5MoeSparseMoeBlock_77627238908541
// MI455X (gfx1250) — matrix operand order repaired and checked by reading, not compiled, not run on hardware
//
#include <hip/hip_runtime.h>
#include <hip/hip_bf16.h>
#include <stdint.h>

#define H_DIM 2048
#define F_DIM 768
#define E_NUM 32
#define TOPK  4
#define SF_DIM 5632
#define T_TOK 2048
#define SF_CHUNK 704                    // 5632 / 8, multiple of 64
#define N_SFCHUNKS (SF_DIM / SF_CHUNK)  // 8
#define MTILE 32
#define TILES_T (T_TOK / MTILE)         // 64

typedef __bf16   v16bf __attribute__((ext_vector_type(16)));
typedef __bf16   v2bf  __attribute__((ext_vector_type(2)));
typedef float    v8f   __attribute__((ext_vector_type(8)));
typedef uint32_t v8u   __attribute__((ext_vector_type(8)));
typedef uint32_t v4u   __attribute__((ext_vector_type(4)));

__device__ __forceinline__ uint16_t f2bf(float f) {
  __bf16 h = (__bf16)f;
  return __builtin_bit_cast(uint16_t, h);
}
// Packed 2xf32 -> 2xbf16 in a single VGPR (v_cvt_pk_bf16_f32 when available).
__device__ __forceinline__ uint32_t pack2bf(float a, float b) {
#if __has_builtin(__builtin_amdgcn_cvt_pk_bf16_f32)
  v2bf r = __builtin_amdgcn_cvt_pk_bf16_f32(a, b);
  return __builtin_bit_cast(uint32_t, r);
#else
  v2bf r = { (__bf16)a, (__bf16)b };
  return __builtin_bit_cast(uint32_t, r);
#endif
}

// 16-bit A-matrix 16x32 K offsets (ISA 7.12.2): VGPR v<4: K=2v+8*half ; v>=4: K=16+2(v-4)+8*half
// => VGPRs 0-3 = 16 contiguous bytes at half-offset k0+8*half, VGPRs 4-7 at k0+16+8*half.
// 16-bit B-matrix 32x16: same K order as A: VGPR v<4: K=2v+8*half ; v>=4: K=16+2(v-4)+8*half.
__device__ __forceinline__ int kbB(int v, int half) { return (v < 4 ? 2 * v : 16 + 2 * (v - 4)) + 8 * half; }

// A fragment: two 128-bit loads (works for global or LDS pointers).
__device__ __forceinline__ v16bf load_a(const uint16_t* row, int k0, int half) {
  v4u lo = *(const v4u*)(row + k0 + 8 * half);        // 16B aligned
  v4u hi = *(const v4u*)(row + k0 + 16 + 8 * half);   // 16B aligned
  v8u u = { lo[0], lo[1], lo[2], lo[3], hi[0], hi[1], hi[2], hi[3] };
  return __builtin_bit_cast(v16bf, u);
}

// B fragment straight from global f32, row-major [K][ldn], column n, rows k0..k0+31.
__device__ __forceinline__ v16bf load_b_gl(const float* __restrict__ W, int ldn, int n,
                                           int k0, int half) {
  v8u u;
#pragma unroll
  for (int v = 0; v < 8; ++v) {
    int k = k0 + kbB(v, half);
    const float* p = W + (size_t)k * ldn + n;
    u[v] = pack2bf(p[0], p[ldn]);                     // (k,n) and (k+1,n)
  }
  return __builtin_bit_cast(v16bf, u);
}

// ---------------------------------------------------------------- kernels ---

__global__ void k_prep(const float* __restrict__ x, uint16_t* __restrict__ xbf,
                       float* __restrict__ out, int n) {
  int i = blockIdx.x * blockDim.x + threadIdx.x;
  if (i < n) { xbf[i] = f2bf(x[i]); out[i] = 0.0f; }
}

// One wave32 per token: lane == expert. Softmax, top-4 (tie -> lower lane),
// normalized weights, plus sigmoid(x . seg) for the shared-expert gate.
__global__ void k_router(const float* __restrict__ x, const float* __restrict__ gw,
                         const float* __restrict__ seg, int* __restrict__ topi,
                         float* __restrict__ topw, float* __restrict__ shg) {
  int t = blockIdx.x;
  int lane = threadIdx.x;                   // 0..31
  const float* xr = x + (size_t)t * H_DIM;
  float logit = 0.0f, segacc = 0.0f;
  for (int h = 0; h < H_DIM; ++h) logit = fmaf(xr[h], gw[h * E_NUM + lane], logit);
  for (int h = lane; h < H_DIM; h += 32) segacc = fmaf(xr[h], seg[h], segacc);
  for (int o = 16; o > 0; o >>= 1) segacc += __shfl_xor(segacc, o, 32);

  float mx = logit;
  for (int o = 16; o > 0; o >>= 1) mx = fmaxf(mx, __shfl_xor(mx, o, 32));
  float p = __expf(logit - mx);
  float s = p;
  for (int o = 16; o > 0; o >>= 1) s += __shfl_xor(s, o, 32);
  float prob = p / s;

  float selw[TOPK]; int seli[TOPK];
  float cur = prob;
#pragma unroll
  for (int k = 0; k < TOPK; ++k) {
    float bv = cur; int bi = lane;
    for (int o = 16; o > 0; o >>= 1) {
      float ov = __shfl_xor(bv, o, 32);
      int   oi = __shfl_xor(bi, o, 32);
      if (ov > bv || (ov == bv && oi < bi)) { bv = ov; bi = oi; }
    }
    selw[k] = bv; seli[k] = bi;
    if (lane == bi) cur = -1.0f;
  }
  if (lane == 0) {
    float tot = selw[0] + selw[1] + selw[2] + selw[3];
#pragma unroll
    for (int k = 0; k < TOPK; ++k) {
      topi[t * TOPK + k] = seli[k];
      topw[t * TOPK + k] = selw[k] / tot;
    }
    shg[t] = 1.0f / (1.0f + __expf(-segacc));
  }
}

__global__ void k_build(const int* __restrict__ topi, const float* __restrict__ topw,
                        int* __restrict__ toklist, float* __restrict__ tokw,
                        int* __restrict__ counts) {
  int e = blockIdx.x;
  __shared__ int cnt;
  if (threadIdx.x == 0) cnt = 0;
  __syncthreads();
  for (int t = threadIdx.x; t < T_TOK; t += blockDim.x) {
#pragma unroll
    for (int k = 0; k < TOPK; ++k) {
      if (topi[t * TOPK + k] == e) {
        int p = atomicAdd(&cnt, 1);
        toklist[e * T_TOK + p] = t;
        tokw[e * T_TOK + p]    = topw[t * TOPK + k];
      }
    }
  }
  __syncthreads();
  if (threadIdx.x == 0) counts[e] = cnt;
}

// Core FFN tile: 32 gathered tokens, block = 128 threads (4 waves), wave owns a
// 16-col slice of each 64-col group; each B fragment feeds 2 WMMAs (M reuse).
// k-loops are software-pipelined: fragments for k+32 are in flight while the
// XDL pipe chews on k.
// Phase 1: h[32][Fchunk] = silu(x@W1) * (x@W2)  (f32 WMMA accum -> bf16 in LDS)
// Phase 2: out[tok] += w * (h @ Wd)             (atomic f32 add)
__device__ __forceinline__ void ffn_tile_core(
    const uint16_t* __restrict__ xbf, const float* __restrict__ W1,
    const float* __restrict__ W2, int ldn1, const float* __restrict__ Wd,
    int Fchunk, float* __restrict__ out, const int* tokL, const float* wL,
    uint16_t* ldsH) {
  const int tid  = threadIdx.x;
  const int lane = tid & 31;
  const int wave = tid >> 5;
  const int m    = lane & 15;
  const int half = lane >> 4;

  const uint16_t* xrow0 = xbf + (size_t)tokL[m]      * H_DIM;  // A tile 0 row
  const uint16_t* xrow1 = xbf + (size_t)tokL[m + 16] * H_DIM;  // A tile 1 row

  // ---- Phase 1: gate/up GEMMs + SwiGLU ----
  const int nGroups1 = Fchunk / 64;
  for (int g = 0; g < nGroups1; ++g) {
    int n0 = g * 64 + wave * 16;
    v8f aG0 = {}, aU0 = {}, aG1 = {}, aU1 = {};
    v16bf a0 = load_a(xrow0, 0, half);
    v16bf a1 = load_a(xrow1, 0, half);
    v16bf bg = load_b_gl(W1, ldn1, n0 + m, 0, half);
    v16bf bu = load_b_gl(W2, ldn1, n0 + m, 0, half);
    for (int k0 = 0; k0 < H_DIM - 32; k0 += 32) {
      v16bf na0 = load_a(xrow0, k0 + 32, half);
      v16bf na1 = load_a(xrow1, k0 + 32, half);
      v16bf nbg = load_b_gl(W1, ldn1, n0 + m, k0 + 32, half);
      v16bf nbu = load_b_gl(W2, ldn1, n0 + m, k0 + 32, half);
      aG0 = __builtin_amdgcn_wmma_f32_16x16x32_bf16(false, a0, false, bg, (short)0, aG0, false, false);
      aG1 = __builtin_amdgcn_wmma_f32_16x16x32_bf16(false, a1, false, bg, (short)0, aG1, false, false);
      aU0 = __builtin_amdgcn_wmma_f32_16x16x32_bf16(false, a0, false, bu, (short)0, aU0, false, false);
      aU1 = __builtin_amdgcn_wmma_f32_16x16x32_bf16(false, a1, false, bu, (short)0, aU1, false, false);
      a0 = na0; a1 = na1; bg = nbg; bu = nbu;
    }
    aG0 = __builtin_amdgcn_wmma_f32_16x16x32_bf16(false, a0, false, bg, (short)0, aG0, false, false);
    aG1 = __builtin_amdgcn_wmma_f32_16x16x32_bf16(false, a1, false, bg, (short)0, aG1, false, false);
    aU0 = __builtin_amdgcn_wmma_f32_16x16x32_bf16(false, a0, false, bu, (short)0, aU0, false, false);
    aU1 = __builtin_amdgcn_wmma_f32_16x16x32_bf16(false, a1, false, bu, (short)0, aU1, false, false);
#pragma unroll
    for (int r = 0; r < 8; ++r) {
      int row0 = r + 8 * half;
      float g0 = aG0[r];
      float s0 = __builtin_amdgcn_rcpf(1.0f + __expf(-g0));    // fast sigmoid
      ldsH[row0 * Fchunk + n0 + m] = f2bf(g0 * s0 * aU0[r]);
      int row1 = 16 + row0;
      float g1 = aG1[r];
      float s1 = __builtin_amdgcn_rcpf(1.0f + __expf(-g1));
      ldsH[row1 * Fchunk + n0 + m] = f2bf(g1 * s1 * aU1[r]);
    }
  }
  __syncthreads();

  // ---- Phase 2: down-projection, scale, atomic combine ----
  const uint16_t* hrow0 = ldsH + m * Fchunk;
  const uint16_t* hrow1 = ldsH + (m + 16) * Fchunk;
  const int nGroups2 = H_DIM / 64;
  for (int g = 0; g < nGroups2; ++g) {
    int n0 = g * 64 + wave * 16;
    v8f c0 = {}, c1 = {};
    v16bf a0 = load_a(hrow0, 0, half);
    v16bf a1 = load_a(hrow1, 0, half);
    v16bf b  = load_b_gl(Wd, H_DIM, n0 + m, 0, half);
    for (int k0 = 0; k0 < Fchunk - 32; k0 += 32) {
      v16bf na0 = load_a(hrow0, k0 + 32, half);
      v16bf na1 = load_a(hrow1, k0 + 32, half);
      v16bf nb  = load_b_gl(Wd, H_DIM, n0 + m, k0 + 32, half);
      c0 = __builtin_amdgcn_wmma_f32_16x16x32_bf16(false, a0, false, b, (short)0, c0, false, false);
      c1 = __builtin_amdgcn_wmma_f32_16x16x32_bf16(false, a1, false, b, (short)0, c1, false, false);
      a0 = na0; a1 = na1; b = nb;
    }
    c0 = __builtin_amdgcn_wmma_f32_16x16x32_bf16(false, a0, false, b, (short)0, c0, false, false);
    c1 = __builtin_amdgcn_wmma_f32_16x16x32_bf16(false, a1, false, b, (short)0, c1, false, false);
#pragma unroll
    for (int r = 0; r < 8; ++r) {
      int row0 = r + 8 * half;
      atomicAdd(out + (size_t)tokL[row0] * H_DIM + n0 + m, c0[r] * wL[row0]);
      int row1 = 16 + row0;
      atomicAdd(out + (size_t)tokL[row1] * H_DIM + n0 + m, c1[r] * wL[row1]);
    }
  }
}

#define LDS_H_BYTES (MTILE * F_DIM * 2)   // 48 KB (>= MTILE * SF_CHUNK * 2)

__global__ void k_expert_ffn(const uint16_t* __restrict__ xbf,
                             const float* __restrict__ wg, const float* __restrict__ wu,
                             const float* __restrict__ wd, const int* __restrict__ toklist,
                             const float* __restrict__ tokw, const int* __restrict__ counts,
                             float* __restrict__ out) {
  int e    = blockIdx.x >> 6;         // / 64
  int tile = blockIdx.x & 63;
  int cnt  = counts[e];
  if (tile * MTILE >= cnt) return;    // block-uniform

  extern __shared__ char smem[];
  uint16_t* ldsH = (uint16_t*)smem;
  int*   tokL    = (int*)(smem + LDS_H_BYTES);
  float* wL      = (float*)(tokL + MTILE);

  if (threadIdx.x < MTILE) {
    int slot = tile * MTILE + threadIdx.x;
    if (slot < cnt) {
      tokL[threadIdx.x] = toklist[e * T_TOK + slot];
      wL[threadIdx.x]   = tokw[e * T_TOK + slot];
    } else {                                   // pad with weight 0 (adds exact 0)
      tokL[threadIdx.x] = toklist[e * T_TOK];
      wL[threadIdx.x]   = 0.0f;
    }
  }
  __syncthreads();

  const float* W1 = wg + (size_t)e * H_DIM * F_DIM;
  const float* W2 = wu + (size_t)e * H_DIM * F_DIM;
  const float* Wd = wd + (size_t)e * F_DIM * H_DIM;
  ffn_tile_core(xbf, W1, W2, F_DIM, Wd, F_DIM, out, tokL, wL, ldsH);
}

__global__ void k_shared_ffn(const uint16_t* __restrict__ xbf,
                             const float* __restrict__ sg, const float* __restrict__ su,
                             const float* __restrict__ sd, const float* __restrict__ shg,
                             float* __restrict__ out) {
  int tile  = blockIdx.x >> 3;   // 0..63
  int chunk = blockIdx.x & 7;    // 0..7

  extern __shared__ char smem[];
  uint16_t* ldsH = (uint16_t*)smem;
  int*   tokL    = (int*)(smem + LDS_H_BYTES);
  float* wL      = (float*)(tokL + MTILE);

  if (threadIdx.x < MTILE) {
    int t = tile * MTILE + threadIdx.x;
    tokL[threadIdx.x] = t;
    wL[threadIdx.x]   = shg[t];
  }
  __syncthreads();

  const float* W1 = sg + chunk * SF_CHUNK;                 // column slice, ldn = SF
  const float* W2 = su + chunk * SF_CHUNK;
  const float* Wd = sd + (size_t)(chunk * SF_CHUNK) * H_DIM;
  ffn_tile_core(xbf, W1, W2, SF_DIM, Wd, SF_CHUNK, out, tokL, wL, ldsH);
}

// ---------------------------------------------------------------- launch ----

extern "C" void kernel_launch(void* const* d_in, const int* in_sizes, int n_in,
                              void* d_out, int out_size, void* d_ws, size_t ws_size,
                              hipStream_t stream) {
  const float* x   = (const float*)d_in[0];   // [T, H]
  const float* gw  = (const float*)d_in[1];   // [H, E]
  const float* wg  = (const float*)d_in[2];   // [E, H, F]
  const float* wu  = (const float*)d_in[3];   // [E, H, F]
  const float* wd  = (const float*)d_in[4];   // [E, F, H]
  const float* sg  = (const float*)d_in[5];   // [H, SF]
  const float* su  = (const float*)d_in[6];   // [H, SF]
  const float* sd  = (const float*)d_in[7];   // [SF, H]
  const float* seg = (const float*)d_in[8];   // [H, 1]
  float* out = (float*)d_out;                 // [T, H]

  char* ws = (char*)d_ws;
  uint16_t* xbf    = (uint16_t*)ws;  ws += (size_t)T_TOK * H_DIM * 2;   // 8 MB
  int*      topi   = (int*)ws;       ws += (size_t)T_TOK * TOPK * 4;
  float*    topw   = (float*)ws;     ws += (size_t)T_TOK * TOPK * 4;
  float*    shg    = (float*)ws;     ws += (size_t)T_TOK * 4;
  int*      counts = (int*)ws;       ws += (size_t)E_NUM * 4;
  int*      toklist= (int*)ws;       ws += (size_t)E_NUM * T_TOK * 4;
  float*    tokw   = (float*)ws;     ws += (size_t)E_NUM * T_TOK * 4;

  int n = T_TOK * H_DIM;
  k_prep<<<(n + 255) / 256, 256, 0, stream>>>(x, xbf, out, n);
  k_router<<<T_TOK, 32, 0, stream>>>(x, gw, seg, topi, topw, shg);
  k_build<<<E_NUM, 256, 0, stream>>>(topi, topw, toklist, tokw, counts);

  size_t lds_bytes = (size_t)LDS_H_BYTES + MTILE * 4 + MTILE * 4;  // ~48.25 KB
  k_expert_ffn<<<E_NUM * TILES_T, 128, lds_bytes, stream>>>(
      xbf, wg, wu, wd, toklist, tokw, counts, out);
  k_shared_ffn<<<TILES_T * N_SFCHUNKS, 128, lds_bytes, stream>>>(
      xbf, sg, su, sd, shg, out);
}
